// EdgeMLP_89378269430354
// MI455X (gfx1250) — compile-verified, matrix operand order checked by reading, not run on hardware
//
#include <hip/hip_runtime.h>
#include <hip/hip_bf16.h>

typedef __attribute__((ext_vector_type(16))) _Float16 v16h;
typedef __attribute__((ext_vector_type(8)))  _Float16 v8h;
typedef __attribute__((ext_vector_type(8)))  float    v8f;

#define D_IN   64
#define H0     256      // 4*D concat width
#define H1     128
#define TILE_E 128

// ---------------------------------------------------------------------------
// Prep: convert W1 [128x256] and W2 [128x128] fp32 -> f16 into workspace.
// ---------------------------------------------------------------------------
__global__ void EdgeMLP_prep_kernel(const float* __restrict__ W1,
                                    const float* __restrict__ W2,
                                    _Float16* __restrict__ W1h,
                                    _Float16* __restrict__ W2h) {
    int t = blockIdx.x * blockDim.x + threadIdx.x;
    if (t < H1 * H0) W1h[t] = (_Float16)W1[t];
    if (t < H1 * H1) W2h[t] = (_Float16)W2[t];
}

// ---------------------------------------------------------------------------
// Main fused edge-MLP kernel. One block = 128 edges, 256 threads (8 wave32).
// Wave w owns output-neuron columns [16w, 16w+16).
// ---------------------------------------------------------------------------
__global__ __launch_bounds__(256)
void EdgeMLP_89378269430354_kernel(const float* __restrict__ h1,
                                   const float* __restrict__ h2,
                                   const int* __restrict__ src,
                                   const int* __restrict__ dst,
                                   const _Float16* __restrict__ W1h,
                                   const float* __restrict__ b1,
                                   const _Float16* __restrict__ W2h,
                                   const float* __restrict__ b2,
                                   const float* __restrict__ W3,
                                   const float* __restrict__ b3,
                                   float* __restrict__ out, int E) {
    extern __shared__ char smem[];
    _Float16* Xs = (_Float16*)smem;                               // [128][256] f16
    _Float16* Hs = (_Float16*)(smem + TILE_E * H0 * 2);           // [128][128] f16
    float* scoreBuf = (float*)(smem + TILE_E * H0 * 2 + TILE_E * H1 * 2); // [128] f32

    const int tid  = threadIdx.x;
    const int e0   = blockIdx.x * TILE_E;
    const int wave = tid >> 5;
    const int lane = tid & 31;
    const int ln   = lane & 15;   // n (B/C/D) or m (A) within 16
    const int hi   = lane >> 4;   // lane-half selects K sub-range / M+8
    const int ncol = wave * 16 + ln;  // this lane's output neuron column

    if (tid < TILE_E) scoreBuf[tid] = b3[0];

    // ---- Gather + fp32->f16 convert into Xs. 2 threads per edge row. ----
    {
        int m    = tid >> 1;     // edge within tile
        int half = tid & 1;      // 0: cols [0,128) = h1[src],h1[dst]
                                 // 1: cols [128,256) = h2[src],h2[src] (per reference)
        int e = e0 + m;
        if (e >= E) e = E - 1;   // clamp (E % 128 == 0 for this problem anyway)
        const int s = src[e];
        const int d = dst[e];
        const float* segA = half ? (h2 + (size_t)s * D_IN) : (h1 + (size_t)s * D_IN);
        const float* segB = half ? (h2 + (size_t)s * D_IN) : (h1 + (size_t)d * D_IN);
        _Float16* row = Xs + m * H0 + half * 128;
#pragma unroll
        for (int i = 0; i < D_IN; i += 4) {
            float4 va = *(const float4*)(segA + i);
            row[i + 0] = (_Float16)va.x;
            row[i + 1] = (_Float16)va.y;
            row[i + 2] = (_Float16)va.z;
            row[i + 3] = (_Float16)va.w;
        }
#pragma unroll
        for (int i = 0; i < D_IN; i += 4) {
            float4 vb = *(const float4*)(segB + i);
            row[64 + i + 0] = (_Float16)vb.x;
            row[64 + i + 1] = (_Float16)vb.y;
            row[64 + i + 2] = (_Float16)vb.z;
            row[64 + i + 3] = (_Float16)vb.w;
        }
    }

    // ---- Preload this wave's W1/W2 B-fragments into VGPRs (resident). ----
    // 16-bit fragment layout: lane-lo holds K = {0..7, 16..23}, lane-hi K+8.
    v16h bw1[8];
#pragma unroll
    for (int ks = 0; ks < 8; ++ks) {
        const _Float16* p = W1h + (size_t)ncol * H0 + ks * 32 + hi * 8;
        v8h lo = *(const v8h*)p;
        v8h hp = *(const v8h*)(p + 16);
        bw1[ks] = __builtin_shufflevector(lo, hp, 0, 1, 2, 3, 4, 5, 6, 7,
                                          8, 9, 10, 11, 12, 13, 14, 15);
    }
    v16h bw2[4];
#pragma unroll
    for (int ks = 0; ks < 4; ++ks) {
        const _Float16* p = W2h + (size_t)ncol * H1 + ks * 32 + hi * 8;
        v8h lo = *(const v8h*)p;
        v8h hp = *(const v8h*)(p + 16);
        bw2[ks] = __builtin_shufflevector(lo, hp, 0, 1, 2, 3, 4, 5, 6, 7,
                                          8, 9, 10, 11, 12, 13, 14, 15);
    }
    const float bias1 = b1[ncol];
    const float bias2 = b2[ncol];
    const float w3n   = W3[ncol];

    __syncthreads();

    // ---- Layer 1: [128x256] @ [256x128]^T via v_wmma_f32_16x16x32_f16 ----
#pragma unroll
    for (int mt = 0; mt < 8; ++mt) {
        v8f acc = {};
#pragma unroll
        for (int ks = 0; ks < 8; ++ks) {
            const _Float16* p = Xs + (mt * 16 + ln) * H0 + ks * 32 + hi * 8;
            v8h lo = *(const v8h*)p;
            v8h hp = *(const v8h*)(p + 16);
            v16h a = __builtin_shufflevector(lo, hp, 0, 1, 2, 3, 4, 5, 6, 7,
                                             8, 9, 10, 11, 12, 13, 14, 15);
            acc = __builtin_amdgcn_wmma_f32_16x16x32_f16(
                false, a, false, bw1[ks], (short)0, acc, false, false);
        }
        // bias + sigmoid, store f16 activation. D layout: m = r + 8*hi.
#pragma unroll
        for (int r = 0; r < 8; ++r) {
            float x = acc[r] + bias1;
            float s = 1.0f / (1.0f + __expf(-x));
            Hs[(mt * 16 + r + hi * 8) * H1 + ncol] = (_Float16)s;
        }
    }
    __syncthreads();

    // ---- Layer 2 (+ fused layer 3 reduction) ----
#pragma unroll
    for (int mt = 0; mt < 8; ++mt) {
        v8f acc = {};
#pragma unroll
        for (int ks = 0; ks < 4; ++ks) {
            const _Float16* p = Hs + (mt * 16 + ln) * H1 + ks * 32 + hi * 8;
            v8h lo = *(const v8h*)p;
            v8h hp = *(const v8h*)(p + 16);
            v16h a = __builtin_shufflevector(lo, hp, 0, 1, 2, 3, 4, 5, 6, 7,
                                             8, 9, 10, 11, 12, 13, 14, 15);
            acc = __builtin_amdgcn_wmma_f32_16x16x32_f16(
                false, a, false, bw2[ks], (short)0, acc, false, false);
        }
        // sigmoid(h2) * W3[n]; tree-reduce over the 16 lanes of each half
        // (shfl_xor with masks 1,2,4,8 never crosses the lane-16 boundary).
#pragma unroll
        for (int r = 0; r < 8; ++r) {
            float x = acc[r] + bias2;
            float s = 1.0f / (1.0f + __expf(-x));
            float p = s * w3n;
            p += __shfl_xor(p, 1, 32);
            p += __shfl_xor(p, 2, 32);
            p += __shfl_xor(p, 4, 32);
            p += __shfl_xor(p, 8, 32);
            if (ln == 0) atomicAdd(&scoreBuf[mt * 16 + r + hi * 8], p);
        }
    }
    __syncthreads();

    if (tid < TILE_E) {
        int e = e0 + tid;
        if (e < E) out[e] = scoreBuf[tid];
    }
}

// ---------------------------------------------------------------------------
extern "C" void kernel_launch(void* const* d_in, const int* in_sizes, int n_in,
                              void* d_out, int out_size, void* d_ws, size_t ws_size,
                              hipStream_t stream) {
    const float* h1 = (const float*)d_in[0];
    const float* h2 = (const float*)d_in[1];
    const int*  src = (const int*)d_in[2];
    const int*  dst = (const int*)d_in[3];
    const float* W1 = (const float*)d_in[4];
    const float* b1 = (const float*)d_in[5];
    const float* W2 = (const float*)d_in[6];
    const float* b2 = (const float*)d_in[7];
    const float* W3 = (const float*)d_in[8];
    const float* b3 = (const float*)d_in[9];
    float* out = (float*)d_out;
    const int E = in_sizes[2];

    _Float16* W1h = (_Float16*)d_ws;          // 128*256 f16 = 64 KB
    _Float16* W2h = W1h + H1 * H0;            // 128*128 f16 = 32 KB

    EdgeMLP_prep_kernel<<<(H1 * H0 + 255) / 256, 256, 0, stream>>>(W1, W2, W1h, W2h);

    const int nblocks = (E + TILE_E - 1) / TILE_E;
    const size_t shmem = (size_t)TILE_E * H0 * 2   // Xs
                       + (size_t)TILE_E * H1 * 2   // Hs
                       + (size_t)TILE_E * 4;       // scoreBuf
    EdgeMLP_89378269430354_kernel<<<nblocks, 256, shmem, stream>>>(
        h1, h2, src, dst, W1h, b1, W2h, b2, W3, b3, out, E);
}
